// DGCNN_43911745634410
// MI455X (gfx1250) — matrix operand order repaired and checked by reading, not compiled, not run on hardware
//
#include <hip/hip_runtime.h>
#include <math.h>

typedef __attribute__((ext_vector_type(16))) _Float16 v16h;
typedef __attribute__((ext_vector_type(8)))  float    v8f;
typedef __attribute__((ext_vector_type(4)))  float    v4f;   // true vector type for NT loads

#define N_NODES 129
#define F_DIM   262144
#define E_EDGES 16384
#define CHUNKS  64
#define CSIZE   (F_DIM / CHUNKS)   // 4096 elements per chunk

// ---------------------------------------------------------------------------
// Kernel 1: the only bandwidth-relevant op. partial[row][chunk] = dot of a
// 4096-element slice of x[row,:] with gcn1_W. x is streamed once (135 MB) ->
// non-temporal b128 loads; W (1 MB) is reused by all 129 rows -> normal loads.
// ---------------------------------------------------------------------------
__global__ void k_matvec_partial(const float* __restrict__ x,
                                 const float* __restrict__ W,
                                 float* __restrict__ partial) {
    __shared__ float red[256];
    const int chunk = blockIdx.x;
    const int row   = blockIdx.y;
    const int tid   = threadIdx.x;
    const v4f* x4 = (const v4f*)(x + (size_t)row * F_DIM + (size_t)chunk * CSIZE);
    const v4f* w4 = (const v4f*)(W + (size_t)chunk * CSIZE);
    float acc = 0.f;
#pragma unroll
    for (int i = 0; i < CSIZE / 4 / 256; ++i) {       // 4 iterations
        v4f a = __builtin_nontemporal_load(&x4[tid + i * 256]);
        v4f b = w4[tid + i * 256];
        acc = fmaf(a.x, b.x, acc);
        acc = fmaf(a.y, b.y, acc);
        acc = fmaf(a.z, b.z, acc);
        acc = fmaf(a.w, b.w, acc);
    }
    red[tid] = acc;
    __syncthreads();
    for (int s = 128; s > 0; s >>= 1) {
        if (tid < s) red[tid] += red[tid + s];
        __syncthreads();
    }
    if (tid == 0) partial[row * CHUNKS + chunk] = red[0];
}

// ---------------------------------------------------------------------------
// Kernel 2: one block per edge channel. GCN x2 (shared xw!), sort-pool,
// conv1+pool, conv2+pool. All state lives in LDS; edge scatter uses ds_add_f32.
// ---------------------------------------------------------------------------
__global__ void k_channel(const float* __restrict__ partial,
                          const int*   __restrict__ edge_index,
                          const float* __restrict__ edge_attr,
                          const float* __restrict__ gcn1_b,
                          const float* __restrict__ gcn2_W,
                          const float* __restrict__ gcn2_b,
                          const float* __restrict__ conv1_w,
                          const float* __restrict__ conv1_b,
                          const float* __restrict__ conv2_w,
                          const float* __restrict__ conv2_b,
                          float* __restrict__ allx) {
    const int c   = blockIdx.x;
    const int tid = threadIdx.x;
    __shared__ float xw_s[N_NODES], deg_s[N_NODES], dinv_s[N_NODES];
    __shared__ float h1_s[N_NODES], h2_s[N_NODES];
    __shared__ float s1_s[N_NODES], s2_s[N_NODES];
    __shared__ float t1_s[3 * 127], p1_s[3 * 125], t2_s[123];

    // reduce mat-vec partials (8256 floats; trivial) + init accumulators
    for (int n = tid; n < N_NODES; n += blockDim.x) {
        float s = 0.f;
        for (int k = 0; k < CHUNKS; ++k) s += partial[n * CHUNKS + k];
        xw_s[n]  = s;
        deg_s[n] = 1.0f;          // self-loop weight
        h1_s[n]  = 0.f;
        h2_s[n]  = 0.f;
    }
    __syncthreads();

    const int* rowp = edge_index;
    const int* colp = edge_index + E_EDGES;

    for (int e = tid; e < E_EDGES; e += blockDim.x)
        atomicAdd(&deg_s[colp[e]], edge_attr[e * 3 + c]);
    __syncthreads();
    for (int n = tid; n < N_NODES; n += blockDim.x) {
        float d = deg_s[n];
        dinv_s[n] = d > 0.f ? rsqrtf(d) : 0.f;
    }
    __syncthreads();

    // GCN layer 1: out[col] += dinv[row]*ew*dinv[col] * xw[row]
    for (int e = tid; e < E_EDGES; e += blockDim.x) {
        int r = rowp[e], cc = colp[e];
        float w = edge_attr[e * 3 + c] * dinv_s[r] * dinv_s[cc];
        atomicAdd(&h1_s[cc], w * xw_s[r]);
    }
    __syncthreads();
    const float b1 = gcn1_b[0], W2 = gcn2_W[0], b2 = gcn2_b[0];
    for (int n = tid; n < N_NODES; n += blockDim.x)
        h1_s[n] = h1_s[n] + dinv_s[n] * dinv_s[n] * xw_s[n] + b1;  // self loop + bias
    __syncthreads();

    // GCN layer 2 (input = h1 * scalar W2, same norm)
    for (int e = tid; e < E_EDGES; e += blockDim.x) {
        int r = rowp[e], cc = colp[e];
        float w = edge_attr[e * 3 + c] * dinv_s[r] * dinv_s[cc];
        atomicAdd(&h2_s[cc], w * h1_s[r] * W2);
    }
    __syncthreads();
    for (int n = tid; n < N_NODES; n += blockDim.x)
        h2_s[n] = h2_s[n] + dinv_s[n] * dinv_s[n] * h1_s[n] * W2 + b2;
    __syncthreads();

    // SortPool: stable descending argsort by h2 (rank via N^2 compare)
    for (int i = tid; i < N_NODES; i += blockDim.x) {
        float vi = h2_s[i];
        int rk = 0;
        for (int j = 0; j < N_NODES; ++j) {
            float vj = h2_s[j];
            rk += (vj > vi) || (vj == vi && j < i);
        }
        s1_s[rk] = h1_s[i];
        s2_s[rk] = vi;
    }
    __syncthreads();

    // conv1: 2->3 channels, k=3, 129 -> 127
    for (int idx = tid; idx < 3 * 127; idx += blockDim.x) {
        int o = idx / 127, j = idx % 127;
        float z = conv1_b[o];
#pragma unroll
        for (int k = 0; k < 3; ++k) {
            z = fmaf(conv1_w[o * 6 + 0 * 3 + k], s1_s[j + k], z);
            z = fmaf(conv1_w[o * 6 + 1 * 3 + k], s2_s[j + k], z);
        }
        t1_s[idx] = z;
    }
    __syncthreads();
    // maxpool3: 127 -> 125
    for (int idx = tid; idx < 3 * 125; idx += blockDim.x) {
        int o = idx / 125, j = idx % 125;
        float m = fmaxf(t1_s[o * 127 + j],
                  fmaxf(t1_s[o * 127 + j + 1], t1_s[o * 127 + j + 2]));
        p1_s[idx] = m;
    }
    __syncthreads();
    // conv2: 3->1, k=3, 125 -> 123
    for (int j = tid; j < 123; j += blockDim.x) {
        float z = conv2_b[0];
#pragma unroll
        for (int i2 = 0; i2 < 3; ++i2)
#pragma unroll
            for (int k = 0; k < 3; ++k)
                z = fmaf(conv2_w[i2 * 3 + k], p1_s[i2 * 125 + j + k], z);
        t2_s[j] = z;
    }
    __syncthreads();
    // maxpool3: 123 -> 121
    for (int j = tid; j < 121; j += blockDim.x)
        allx[c * 121 + j] = fmaxf(t2_s[j], fmaxf(t2_s[j + 1], t2_s[j + 2]));
}

// ---------------------------------------------------------------------------
// Kernel 3: FC head. fc1 (363x32) done with v_wmma_f32_16x16x32_f16 by one
// full wave (EXEC all ones), A/B fragments in contiguous k order (k = 16*(lane/16) + element).
// ---------------------------------------------------------------------------
__device__ __forceinline__ float elu1(float v) { return v > 0.f ? v : (expf(v) - 1.f); }

__global__ void k_head(const float* __restrict__ allx,
                       const float* __restrict__ fc1_W, const float* __restrict__ fc1_b,
                       const float* __restrict__ fc2_W, const float* __restrict__ fc2_b,
                       const float* __restrict__ fc3_W, const float* __restrict__ fc3_b,
                       float* __restrict__ out) {
    __shared__ float a_s[384];     // 363 inputs zero-padded to 12 K-tiles of 32
    __shared__ float fc1_s[32];
    __shared__ float fc2_s[6];
    const int tid = threadIdx.x;   // blockDim.x == 32: exactly one wave
    for (int i = tid; i < 384; i += 32) a_s[i] = (i < 363) ? allx[i] : 0.f;
    __syncthreads();

    const int L = tid;
    const int M = L & 15;
    union U16 { v16h v; _Float16 e[16]; };
    v8f acc0 = {0.f, 0.f, 0.f, 0.f, 0.f, 0.f, 0.f, 0.f};
    v8f acc1 = {0.f, 0.f, 0.f, 0.f, 0.f, 0.f, 0.f, 0.f};

    for (int kt = 0; kt < 12; ++kt) {
        U16 af, bf0, bf1;
#pragma unroll
        for (int h = 0; h < 16; ++h) {
            // A 16x32 f16 layout: lane -> M = L%16; K = (L<16?0:16)+h
            int KA  = ((L < 16) ? 0 : 16) + h;
            int gka = kt * 32 + KA;
            af.e[h] = (_Float16)((M == 0) ? a_s[gka] : 0.f);   // only row 0 is live
            // B 32x16 f16 layout: lane -> N = L%16; K = (L<16?0:16)+h
            int KB  = ((L < 16) ? 0 : 16) + h;
            int gkb = kt * 32 + KB;
            float w0 = (gkb < 363) ? fc1_W[gkb * 32 + M]      : 0.f;
            float w1 = (gkb < 363) ? fc1_W[gkb * 32 + 16 + M] : 0.f;
            bf0.e[h] = (_Float16)w0;
            bf1.e[h] = (_Float16)w1;
        }
        acc0 = __builtin_amdgcn_wmma_f32_16x16x32_f16(false, af.v, false, bf0.v,
                                                      (short)0, acc0, false, false);
        acc1 = __builtin_amdgcn_wmma_f32_16x16x32_f16(false, af.v, false, bf1.v,
                                                      (short)0, acc1, false, false);
    }
    // D layout: M=0 row lives in VGPR0 of lanes 0..15 (N = lane)
    if (L < 16) {
        fc1_s[L]      = acc0[0];
        fc1_s[16 + L] = acc1[0];
    }
    __syncthreads();
    fc1_s[tid] = elu1(fc1_s[tid] + fc1_b[tid]);
    __syncthreads();
    if (tid < 6) {
        float z = fc2_b[tid];
        for (int i = 0; i < 32; ++i) z = fmaf(fc1_s[i], fc2_W[i * 6 + tid], z);
        fc2_s[tid] = elu1(z);
    }
    __syncthreads();
    if (tid < 2) {
        float z = fc3_b[tid];
        for (int i = 0; i < 6; ++i) z = fmaf(fc2_s[i], fc3_W[i * 2 + tid], z);
        out[tid] = z;
    }
}

// ---------------------------------------------------------------------------
extern "C" void kernel_launch(void* const* d_in, const int* in_sizes, int n_in,
                              void* d_out, int out_size, void* d_ws, size_t ws_size,
                              hipStream_t stream) {
    const float* x         = (const float*)d_in[0];
    const int*   edge_index= (const int*)  d_in[1];
    const float* edge_attr = (const float*)d_in[2];
    const float* gcn1_W    = (const float*)d_in[3];
    const float* gcn1_b    = (const float*)d_in[4];
    const float* gcn2_W    = (const float*)d_in[5];
    const float* gcn2_b    = (const float*)d_in[6];
    const float* conv1_w   = (const float*)d_in[7];
    const float* conv1_b   = (const float*)d_in[8];
    const float* conv2_w   = (const float*)d_in[9];
    const float* conv2_b   = (const float*)d_in[10];
    const float* fc1_W     = (const float*)d_in[11];
    const float* fc1_b     = (const float*)d_in[12];
    const float* fc2_W     = (const float*)d_in[13];
    const float* fc2_b     = (const float*)d_in[14];
    const float* fc3_W     = (const float*)d_in[15];
    const float* fc3_b     = (const float*)d_in[16];

    float* partial = (float*)d_ws;                   // 129*64 floats
    float* allx    = partial + N_NODES * CHUNKS;     // 363 floats

    dim3 g1(CHUNKS, N_NODES);                        // 8256 blocks: saturate HBM
    k_matvec_partial<<<g1, 256, 0, stream>>>(x, gcn1_W, partial);
    k_channel<<<3, 256, 0, stream>>>(partial, edge_index, edge_attr,
                                     gcn1_b, gcn2_W, gcn2_b,
                                     conv1_w, conv1_b, conv2_w, conv2_b, allx);
    k_head<<<1, 32, 0, stream>>>(allx, fc1_W, fc1_b, fc2_W, fc2_b, fc3_W, fc3_b,
                                 (float*)d_out);
}
